// admmNet_24885040513551
// MI455X (gfx1250) — matrix operand order repaired and checked by reading, not compiled, not run on hardware
//
#include <hip/hip_runtime.h>
#include <hip/hip_bf16.h>

// ---------------------------------------------------------------------------
// ADMM-Net on gfx1250: fused f16-WMMA GEMM + soft-threshold updates.
//   b = (y/MU) @ A                         (M=8192, K=512,  N=1024)
//   per layer: x = (z+u) @ W^T + b; z=soft(x-u); u=u-(x-z)   (K=N=1024)
// GEMMs use v_wmma_f32_16x16x32_f16; W tiles are staged into LDS with the
// CDNA5 async DMA path (global_load_async_to_lds_b128 / ASYNCcnt).
// ---------------------------------------------------------------------------

typedef __attribute__((ext_vector_type(16))) _Float16 v16h;
typedef __attribute__((ext_vector_type(8)))  _Float16 v8h;
typedef __attribute__((ext_vector_type(8)))  float    v8f;

#define ADMM_MU    1.0f
#define ADMM_LAM   0.1f

// Async global->LDS copy of 16 bytes per lane at immediate offset IMM.
// vdst = per-lane LDS byte offset, vaddr = per-lane 64-bit global address.
#define ASYNC_G2L_B128(ldsoff, gptr, IMM)                                   \
    asm volatile("global_load_async_to_lds_b128 %0, %1, off offset:" #IMM   \
                 :: "v"(ldsoff), "v"(gptr) : "memory")
#define WAIT_ASYNC0() asm volatile("s_wait_asynccnt 0x0" ::: "memory")

// ---------------------------------------------------------------------------
// Conversion kernels (run once per launch; tiny vs GEMM cost)
// ---------------------------------------------------------------------------
__global__ void cvt_scale_f32_f16(const float* __restrict__ in,
                                  _Float16* __restrict__ out,
                                  float scale, int n) {
    int i = blockIdx.x * blockDim.x + threadIdx.x;
    if (i < n) out[i] = (_Float16)(in[i] * scale);
}

// in: [R][C] row-major f32  ->  out: [C][R] row-major f16  (i.e. transpose)
__global__ void cvt_transpose_f32_f16(const float* __restrict__ in,
                                      _Float16* __restrict__ out,
                                      int R, int C) {
    int i = blockIdx.x * blockDim.x + threadIdx.x;
    if (i < R * C) {
        int r = i / C, c = i - r * C;
        out[(size_t)c * R + r] = (_Float16)in[i];
    }
}

// ---------------------------------------------------------------------------
// Fused GEMM + ADMM epilogue.
//   Amat: [M][K] f16 row-major (GEMM A)
//   Bmat: [N][K] f16 row-major (acts as B^T: B[k][n] = Bmat[n][k] —
//         contiguous along K, so each lane reads 16 contiguous k, as for A)
// mode 0: x = A@B^T            ; bout=x; z=soft(x); u=z-x; sout=z+u   (layer 1)
// mode 1: x = A@B^T + bvec     ; z=soft(x-u); u=u-(x-z); sout=z+u     (middle)
// mode 2: xout = A@B^T + bvec                                          (last)
// Block: 256 threads (8 waves). Output tile 256(M) x 128(N); each wave owns
// two 16-row strips so every LDS B-fragment feeds two WMMAs. K step = 32.
// ---------------------------------------------------------------------------
__global__ __launch_bounds__(256)
void admm_gemm(const _Float16* __restrict__ Amat,
               const _Float16* __restrict__ Bmat,
               const float*    __restrict__ bvec,
               float*          __restrict__ bout,
               float*          __restrict__ u,
               _Float16*       __restrict__ sout,
               float*          __restrict__ xout,
               int K, int N, int mode) {
    extern __shared__ __align__(128) _Float16 smem[];   // 2 x 8KB W-tiles

    const int tid  = threadIdx.x;
    const int lane = tid & 31;
    const int wv   = tid >> 5;                    // wave 0..7
    const int m0   = blockIdx.y * 256 + wv * 16;  // strip 0; strip 1 = m0+128
    const int n0   = blockIdx.x * 128;

    // W-tile staging: 128 rows x 32 f16 (64B/row); 2 threads/row, 32B each,
    // copied by async DMA (two b128 per thread at +0 / +16).
    const int row  = tid >> 1;
    const int half = tid & 1;
    const unsigned    lds_thr = (unsigned)(row * 64 + half * 32);
    const char* gbase = (const char*)(Bmat + (size_t)(n0 + row) * K) + half * 32;

    // A-fragment bases (contiguous k order used here, same as B):
    //   lanes 0-15 -> M=lane,    K in {0..15}
    //   lanes16-31 -> M=lane-16, K in {16..31}
    const _Float16* arow0 =
        Amat + (size_t)(m0 + (lane & 15)) * K + ((lane >> 4) << 4);
    const _Float16* arow1 = arow0 + (size_t)128 * K;

    v8f acc[2][8];
#pragma unroll
    for (int s = 0; s < 2; ++s)
#pragma unroll
        for (int t = 0; t < 8; ++t)
            acc[s][t] = (v8f){0.f,0.f,0.f,0.f,0.f,0.f,0.f,0.f};

    // Prologue: kick off async copy of the first W tile into buffer 0.
    ASYNC_G2L_B128(lds_thr, gbase, 0);
    ASYNC_G2L_B128(lds_thr, gbase, 16);

    int c = 0;
    for (int k0 = 0; k0 < K; k0 += 32) {
        WAIT_ASYNC0();        // my slice of tile c has landed in LDS
        __syncthreads();      // everyone's slice has landed / readers done

        // Prefetch next W tile into the other buffer (overlaps with compute).
        if (k0 + 32 < K) {
            const char* gnext = gbase + (size_t)2 * (k0 + 32);
            unsigned ldsn = lds_thr + (unsigned)((c ^ 1) * 8192);
            ASYNC_G2L_B128(ldsn, gnext, 0);
            ASYNC_G2L_B128(ldsn, gnext, 16);
            __builtin_prefetch(arow0 + k0 + 64, 0, 3);
        }

        // Two A fragments (two 16B loads each, contiguous k order, same as B).
        v8h lo0 = *(const v8h*)(arow0 + k0);
        v8h hi0 = *(const v8h*)(arow0 + k0 + 8);
        v8h lo1 = *(const v8h*)(arow1 + k0);
        v8h hi1 = *(const v8h*)(arow1 + k0 + 8);
        v16h a0, a1;
#pragma unroll
        for (int i = 0; i < 8; ++i) {
            a0[i] = lo0[i]; a0[8 + i] = hi0[i];
            a1[i] = lo1[i]; a1[8 + i] = hi1[i];
        }

        // 8 N-tiles; each LDS B-fragment feeds both M-strips (2 WMMAs).
        const _Float16* wtc = smem + c * 4096;
#pragma unroll
        for (int t = 0; t < 8; ++t) {
            v16h bb = *(const v16h*)
                (wtc + ((lane & 15) + (t << 4)) * 32 + ((lane >> 4) << 4));
            acc[0][t] = __builtin_amdgcn_wmma_f32_16x16x32_f16(
                false, a0, false, bb, (short)0, acc[0][t], false, false);
            acc[1][t] = __builtin_amdgcn_wmma_f32_16x16x32_f16(
                false, a1, false, bb, (short)0, acc[1][t], false, false);
        }
        c ^= 1;
    }

    // Epilogue. C/D layout: VGPR r, lanes 0-15 -> M=r, lanes 16-31 -> M=r+8.
    const int ncol = n0 + (lane & 15);
    for (int s = 0; s < 2; ++s) {
        const int mrow0 = m0 + s * 128 + ((lane >> 4) << 3);
#pragma unroll
        for (int t = 0; t < 8; ++t) {
            const int n = ncol + (t << 4);
#pragma unroll
            for (int r = 0; r < 8; ++r) {
                const size_t idx = (size_t)(mrow0 + r) * N + n;
                float x = acc[s][t][r];
                if (mode == 0) {
                    bout[idx] = x;                       // b for later layers
                    float a1v = __builtin_fabsf(x) - ADMM_LAM;
                    float zn  = a1v > 0.f ? __builtin_copysignf(a1v, x) : 0.f;
                    float un  = zn - x;                  // u1 = z1 - b
                    u[idx]    = un;
                    sout[idx] = (_Float16)(zn + un);
                } else if (mode == 1) {
                    x += bvec[idx];
                    float uo  = u[idx];
                    float tt  = x - uo;
                    float a1v = __builtin_fabsf(tt) - ADMM_LAM;
                    float zn  = a1v > 0.f ? __builtin_copysignf(a1v, tt) : 0.f;
                    float un  = uo - (x - zn);           // gamma = 1
                    u[idx]    = un;
                    sout[idx] = (_Float16)(zn + un);
                } else {
                    xout[idx] = x + bvec[idx];           // final x
                }
            }
        }
    }
}

// ---------------------------------------------------------------------------
// Host-side orchestration (graph-capture safe: all work on `stream`).
// ---------------------------------------------------------------------------
extern "C" void kernel_launch(void* const* d_in, const int* in_sizes, int n_in,
                              void* d_out, int out_size, void* d_ws, size_t ws_size,
                              hipStream_t stream) {
    constexpr int Bsz = 8192, NY = 512, NX = 1024, L = 16;
    constexpr size_t LDS_BYTES = 2 * 8192;

    const float* y  = (const float*)d_in[0];   // [B, NY]
    const float* A  = (const float*)d_in[1];   // [NY, NX]
    const float* Ws = (const float*)d_in[2];   // [L, NX, NX]
    float* xout = (float*)d_out;               // [B, NX]

    char* p = (char*)d_ws;
    auto alloc = [&](size_t bytes) -> void* {
        void* q = (void*)p;
        p += (bytes + 255) & ~(size_t)255;
        return q;
    };
    _Float16* yh  = (_Float16*)alloc((size_t)Bsz * NY * 2);   //  8 MB
    _Float16* AhT = (_Float16*)alloc((size_t)NX * NY * 2);    //  1 MB (A^T)
    _Float16* Wh  = (_Float16*)alloc((size_t)L * NX * NX * 2);// 32 MB
    float*    bb  = (float*)   alloc((size_t)Bsz * NX * 4);   // 32 MB
    float*    uu  = (float*)   alloc((size_t)Bsz * NX * 4);   // 32 MB
    _Float16* s0  = (_Float16*)alloc((size_t)Bsz * NX * 2);   // 16 MB
    _Float16* s1  = (_Float16*)alloc((size_t)Bsz * NX * 2);   // 16 MB

    // Precision conversions (f32 -> f16), fold 1/MU into y.
    {
        int n = Bsz * NY;
        cvt_scale_f32_f16<<<(n + 255) / 256, 256, 0, stream>>>(
            y, yh, 1.0f / ADMM_MU, n);
    }
    {
        int n = NY * NX;
        cvt_transpose_f32_f16<<<(n + 255) / 256, 256, 0, stream>>>(
            A, AhT, NY, NX);
    }
    {
        int n = L * NX * NX;
        cvt_scale_f32_f16<<<(n + 255) / 256, 256, 0, stream>>>(
            Ws, Wh, 1.0f, n);
    }

    dim3 grid(NX / 128, Bsz / 256);   // 8 x 32 blocks of 256 threads

    // Layer 1: x = b (z0=u0=0) -> produce b, u1, s1 in one fused GEMM.
    admm_gemm<<<grid, 256, LDS_BYTES, stream>>>(yh, AhT, nullptr, bb, uu, s0,
                                                nullptr, NY, NX, 0);

    // Layers 2..L (ping-pong s buffers; last layer writes x to d_out).
    for (int l = 1; l < L; ++l) {
        const _Float16* sin = (l & 1) ? s0 : s1;
        _Float16*       so  = (l & 1) ? s1 : s0;
        int mode = (l == L - 1) ? 2 : 1;
        admm_gemm<<<grid, 256, LDS_BYTES, stream>>>(
            sin, Wh + (size_t)l * NX * NX, bb, nullptr, uu, so, xout,
            NX, NX, mode);
    }
}
